// Cell_Lloc_Pre_6227702579242
// MI455X (gfx1250) — compile-verified, matrix operand order checked by reading, not run on hardware
//
#include <hip/hip_runtime.h>
#include <hip/hip_bf16.h>

// ---------------------------------------------------------------------------
// Types for CDNA5 WMMA (wave32): A/B = 16 bf16 per lane, C/D = 8 f32 per lane
// ---------------------------------------------------------------------------
typedef __bf16 v16bf __attribute__((ext_vector_type(16)));
typedef float  v8f   __attribute__((ext_vector_type(8)));

union Frag16 {
  uint4 u[2];   // 32 bytes = 16 bf16
  v16bf v;
};

__device__ __forceinline__ unsigned short f2bf(float f) {
  union { float f; unsigned u; } v; v.f = f;
  unsigned r = v.u + 0x7FFFu + ((v.u >> 16) & 1u);   // round-to-nearest-even
  return (unsigned short)(r >> 16);
}

__device__ __forceinline__ v8f wmma_bf16(const Frag16& a, const Frag16& b, v8f c) {
  // D = A(16x32) * B(32x16) + C(16x16), f32 accumulate
  return __builtin_amdgcn_wmma_f32_16x16x32_bf16(false, a.v, false, b.v,
                                                 (short)0, c, false, false);
}

__device__ __forceinline__ float activate(float v, int act) {
  if (act == 1) return fmaxf(v, 0.0f);
  if (act == 2) return v > 0.0f ? v : 0.01f * v;
  return v;
}

#define NPAIR 261632   // 512*511 pairs per batch
#define KEDGE 4096     // top-k edges per batch per type

// ---------------------------------------------------------------------------
// 0) NCHW f32 -> NHWC bf16  (input: 4 x 1024 x 48 x 48)
// ---------------------------------------------------------------------------
__global__ void nchw_to_nhwc_bf16(const float* __restrict__ in,
                                  unsigned short* __restrict__ out) {
  size_t o = (size_t)blockIdx.x * 256 + threadIdx.x;
  if (o >= (size_t)9216 * 1024) return;
  int c = (int)(o & 1023);
  size_t m = o >> 10;                 // (b,y,x) flat
  int x = (int)(m % 48);
  int y = (int)((m / 48) % 48);
  int b = (int)(m / 2304);
  out[o] = f2bf(in[(((size_t)b * 1024 + c) * 48 + y) * 48 + x]);
}

// ---------------------------------------------------------------------------
// 1) Weight pre-swizzle into WMMA B-fragment layout.
//    Layout: frag[((s*Ntiles + t)*32 + lane)*2 + part] = 8 bf16, where
//    k = s*32 + part*16 + (lane>>4)*8 + j, n = t*16 + (lane&15)
//    convMode: W value taken from OIHW conv weights (k = (ky*3+kx)*1024 + c)
// ---------------------------------------------------------------------------
__global__ void preswz_kernel(const float* __restrict__ W, uint4* __restrict__ out,
                              int Ksteps, int Ntiles, int K, int N, int convMode) {
  int idx = blockIdx.x * blockDim.x + threadIdx.x;
  int total = Ksteps * Ntiles * 64;
  if (idx >= total) return;
  int part = idx & 1;
  int lane = (idx >> 1) & 31;
  int rest = idx >> 6;
  int t = rest % Ntiles;
  int s = rest / Ntiles;
  int n = t * 16 + (lane & 15);
  int kb = s * 32 + part * 16 + (lane >> 4) * 8;
  unsigned short h[8];
#pragma unroll
  for (int j = 0; j < 8; ++j) {
    int k = kb + j;
    float v = 0.0f;
    if (n < N && k < K) {
      if (convMode) {
        int c = k & 1023, kyx = k >> 10;            // k = (ky*3+kx)*1024 + c
        v = W[(size_t)n * 9216 + (size_t)c * 9 + kyx]; // W_dec OIHW (256,1024,3,3)
      } else {
        v = W[(size_t)k * N + n];                   // row-major K x N
      }
    }
    h[j] = f2bf(v);
  }
  uint4 u;
  u.x = (unsigned)h[0] | ((unsigned)h[1] << 16);
  u.y = (unsigned)h[2] | ((unsigned)h[3] << 16);
  u.z = (unsigned)h[4] | ((unsigned)h[5] << 16);
  u.w = (unsigned)h[6] | ((unsigned)h[7] << 16);
  out[idx] = u;
}

// ---------------------------------------------------------------------------
// 2) Conv 3x3 1024->256 as implicit GEMM, WMMA bf16, fused bias+ReLU.
//    M = 9216 (b,y,x), N = 256, K = 9*1024.
//    Block = 8 waves; wave tile = 32M x 32N (2x2 frags, 4 WMMA/K-step);
//    block tile = 64M x 128N.  Output dec stored NHWC f32 [m][n].
// ---------------------------------------------------------------------------
__global__ __launch_bounds__(256) void conv_wmma_kernel(
    const unsigned short* __restrict__ inb,   // NHWC bf16
    const uint4* __restrict__ Wf,             // pre-swizzled frags (288 x 16 tiles)
    const float* __restrict__ bdec,
    float* __restrict__ dec) {
  const int wid   = threadIdx.x >> 5;
  const int lane  = threadIdx.x & 31;
  const int hl    = lane >> 4;
  const int m0    = blockIdx.x * 64 + (wid >> 2) * 32;
  const int tbase = blockIdx.y * 8 + (wid & 3) * 2;

  // two M rows per lane (m-tile 0 and m-tile 1)
  const int mA = m0 + (lane & 15);
  const int mB = mA + 16;
  const int bA = mA / 2304, remA = mA % 2304, yA = remA / 48, xA = remA % 48;
  const int bB = mB / 2304, remB = mB % 2304, yB = remB / 48, xB = remB % 48;

  v8f acc00 = {0.f,0.f,0.f,0.f,0.f,0.f,0.f,0.f};
  v8f acc01 = acc00, acc10 = acc00, acc11 = acc00;

  for (int kyx = 0; kyx < 9; ++kyx) {
    const int ky = kyx / 3, kx = kyx % 3;
    const int yyA = yA + ky - 1, xxA = xA + kx - 1;
    const int yyB = yB + ky - 1, xxB = xB + kx - 1;
    const bool validA = (yyA >= 0) && (yyA < 48) && (xxA >= 0) && (xxA < 48);
    const bool validB = (yyB >= 0) && (yyB < 48) && (xxB >= 0) && (xxB < 48);
    const unsigned short* rpA =
        inb + ((size_t)((bA * 48 + yyA) * 48 + xxA)) * 1024 + hl * 8;
    const unsigned short* rpB =
        inb + ((size_t)((bB * 48 + yyB) * 48 + xxB)) * 1024 + hl * 8;
#pragma unroll 4
    for (int cs = 0; cs < 32; ++cs) {
      Frag16 fa0, fa1;
      if (validA) {
        const uint4* p = (const uint4*)(rpA + cs * 32);
        fa0.u[0] = p[0];      // K = cs*32 + hl*8 + [0..7]
        fa0.u[1] = p[2];      // K = cs*32 + 16 + hl*8 + [0..7]
      } else {
        fa0.u[0] = make_uint4(0, 0, 0, 0);
        fa0.u[1] = make_uint4(0, 0, 0, 0);
      }
      if (validB) {
        const uint4* p = (const uint4*)(rpB + cs * 32);
        fa1.u[0] = p[0];
        fa1.u[1] = p[2];
      } else {
        fa1.u[0] = make_uint4(0, 0, 0, 0);
        fa1.u[1] = make_uint4(0, 0, 0, 0);
      }
      const int s = kyx * 32 + cs;
      const size_t bi0 = ((size_t)(s * 16 + tbase) * 32 + lane) * 2;
      const size_t bi1 = ((size_t)(s * 16 + tbase + 1) * 32 + lane) * 2;
      Frag16 fb0, fb1;
      fb0.u[0] = Wf[bi0]; fb0.u[1] = Wf[bi0 + 1];
      fb1.u[0] = Wf[bi1]; fb1.u[1] = Wf[bi1 + 1];
      acc00 = wmma_bf16(fa0, fb0, acc00);
      acc01 = wmma_bf16(fa0, fb1, acc01);
      acc10 = wmma_bf16(fa1, fb0, acc10);
      acc11 = wmma_bf16(fa1, fb1, acc11);
    }
  }

#pragma unroll
  for (int r = 0; r < 8; ++r) {
    const int mm0 = m0 + r + hl * 8;
    const int mm1 = mm0 + 16;
    const int n0 = tbase * 16 + (lane & 15);
    const int n1 = n0 + 16;
    const float bn0 = bdec[n0], bn1 = bdec[n1];
    dec[(size_t)mm0 * 256 + n0] = fmaxf(acc00[r] + bn0, 0.0f);
    dec[(size_t)mm0 * 256 + n1] = fmaxf(acc01[r] + bn1, 0.0f);
    dec[(size_t)mm1 * 256 + n0] = fmaxf(acc10[r] + bn0, 0.0f);
    dec[(size_t)mm1 * 256 + n1] = fmaxf(acc11[r] + bn1, 0.0f);
  }
}

// ---------------------------------------------------------------------------
// 3) Generic WMMA GEMM: C = act(A(bf16, MxK row-major) * Bfrag + bias)
//    Block = 8 waves; wave tile = 32M x 32N (2x2); block tile = 64M x 128N.
//    M must be a multiple of 64. Optional f32 / bf16 outputs.
// ---------------------------------------------------------------------------
__global__ __launch_bounds__(256) void gemm_wmma_kernel(
    const unsigned short* __restrict__ A, int lda,
    const uint4* __restrict__ Bf, int Ntiles,
    const float* __restrict__ bias,
    float* __restrict__ outF, unsigned short* __restrict__ outH,
    int ldc, int ncolOff, int N, int Ksteps, int act) {
  const int wid   = threadIdx.x >> 5;
  const int lane  = threadIdx.x & 31;
  const int hl    = lane >> 4;
  const int m0    = blockIdx.x * 64 + (wid >> 2) * 32;
  const int tbase = blockIdx.y * 8 + (wid & 3) * 2;
  const int t0 = tbase, t1 = tbase + 1;
  const bool has0 = t0 < Ntiles, has1 = t1 < Ntiles;

  v8f acc00 = {0.f,0.f,0.f,0.f,0.f,0.f,0.f,0.f};
  v8f acc01 = acc00, acc10 = acc00, acc11 = acc00;

  const unsigned short* ap0 = A + (size_t)(m0 + (lane & 15)) * lda + hl * 8;
  const unsigned short* ap1 = ap0 + (size_t)16 * lda;
#pragma unroll 4
  for (int s = 0; s < Ksteps; ++s) {
    Frag16 fa0, fa1;
    {
      const uint4* p = (const uint4*)(ap0 + s * 32);
      fa0.u[0] = p[0]; fa0.u[1] = p[2];
    }
    {
      const uint4* p = (const uint4*)(ap1 + s * 32);
      fa1.u[0] = p[0]; fa1.u[1] = p[2];
    }
    if (has0) {
      const size_t bi = ((size_t)(s * Ntiles + t0) * 32 + lane) * 2;
      Frag16 fb; fb.u[0] = Bf[bi]; fb.u[1] = Bf[bi + 1];
      acc00 = wmma_bf16(fa0, fb, acc00);
      acc10 = wmma_bf16(fa1, fb, acc10);
    }
    if (has1) {
      const size_t bi = ((size_t)(s * Ntiles + t1) * 32 + lane) * 2;
      Frag16 fb; fb.u[0] = Bf[bi]; fb.u[1] = Bf[bi + 1];
      acc01 = wmma_bf16(fa0, fb, acc01);
      acc11 = wmma_bf16(fa1, fb, acc11);
    }
  }

#pragma unroll
  for (int r = 0; r < 8; ++r) {
    const int mm0 = m0 + r + hl * 8;
    const int mm1 = mm0 + 16;
    if (has0) {
      const int n = t0 * 16 + (lane & 15);
      if (n < N) {
        const float bb = bias ? bias[n] : 0.0f;
        float v0 = activate(acc00[r] + bb, act);
        float v1 = activate(acc10[r] + bb, act);
        size_t o0 = (size_t)mm0 * ldc + ncolOff + n;
        size_t o1 = (size_t)mm1 * ldc + ncolOff + n;
        if (outF) { outF[o0] = v0; outF[o1] = v1; }
        if (outH) { outH[o0] = f2bf(v0); outH[o1] = f2bf(v1); }
      }
    }
    if (has1) {
      const int n = t1 * 16 + (lane & 15);
      if (n < N) {
        const float bb = bias ? bias[n] : 0.0f;
        float v0 = activate(acc01[r] + bb, act);
        float v1 = activate(acc11[r] + bb, act);
        size_t o0 = (size_t)mm0 * ldc + ncolOff + n;
        size_t o1 = (size_t)mm1 * ldc + ncolOff + n;
        if (outF) { outF[o0] = v0; outF[o1] = v1; }
        if (outH) { outH[o0] = f2bf(v0); outH[o1] = f2bf(v1); }
      }
    }
  }
}

// ---------------------------------------------------------------------------
// 4) ROI-align 2x2 over NHWC dec -> cnn bf16 (2048 x 1024), feature = c*4 + s
// ---------------------------------------------------------------------------
__global__ void roi_kernel(const float* __restrict__ boxes,
                           const float* __restrict__ dec,
                           unsigned short* __restrict__ cnnB) {
  const int i = blockIdx.x;          // box 0..2047
  const int c = threadIdx.x;         // channel 0..255
  const int b = i >> 9;
  const float x1 = boxes[i * 4 + 0], y1 = boxes[i * 4 + 1];
  const float x2 = boxes[i * 4 + 2], y2 = boxes[i * 4 + 3];
  const float bw = (x2 - x1) * 0.5f, bh = (y2 - y1) * 0.5f;
  const float sx[2] = { x1 + 0.5f * bw, x1 + 1.5f * bw };
  const float sy[2] = { y1 + 0.5f * bh, y1 + 1.5f * bh };
  const size_t base = (size_t)b * 2304 * 256;
#pragma unroll
  for (int s = 0; s < 4; ++s) {
    float py = fminf(fmaxf(sy[s >> 1], 0.0f), 47.0f);
    float px = fminf(fmaxf(sx[s & 1], 0.0f), 47.0f);
    int y0 = (int)floorf(py), x0 = (int)floorf(px);
    int y1i = min(y0 + 1, 47), x1i = min(x0 + 1, 47);
    float wy = py - (float)y0, wx = px - (float)x0;
    float f00 = dec[base + ((size_t)y0  * 48 + x0 ) * 256 + c];
    float f01 = dec[base + ((size_t)y0  * 48 + x1i) * 256 + c];
    float f10 = dec[base + ((size_t)y1i * 48 + x0 ) * 256 + c];
    float f11 = dec[base + ((size_t)y1i * 48 + x1i) * 256 + c];
    float v = f00 * (1 - wy) * (1 - wx) + f01 * (1 - wy) * wx +
              f10 * wy * (1 - wx) + f11 * wy * wx;
    cnnB[(size_t)i * 1024 + c * 4 + s] = f2bf(v);
  }
}

// ---------------------------------------------------------------------------
// 5) Box features -> fusion[:, 0:256] (bf16), fused ReLU
// ---------------------------------------------------------------------------
__global__ void box_feat_kernel(const float* __restrict__ boxes,
                                const float* __restrict__ W_box,
                                const float* __restrict__ b_box,
                                unsigned short* __restrict__ fusion) {
  const int i = blockIdx.x;
  const int j = threadIdx.x;
  const float x1 = boxes[i * 4 + 0], y1 = boxes[i * 4 + 1];
  const float x2 = boxes[i * 4 + 2], y2 = boxes[i * 4 + 3];
  const float f0 = (x1 + x2) * 0.5f / 48.0f;
  const float f1 = (y1 + y2) * 0.5f / 48.0f;
  const float f2 = (x2 - x1) / 48.0f;
  const float f3 = (y2 - y1) / 48.0f;
  float acc = b_box[j] + f0 * W_box[j] + f1 * W_box[256 + j] +
              f2 * W_box[512 + j] + f3 * W_box[768 + j];
  fusion[(size_t)i * 768 + j] = f2bf(fmaxf(acc, 0.0f));
}

// ---------------------------------------------------------------------------
// 6) Per-batch box centers + table extents (LDS max-reduction)
// ---------------------------------------------------------------------------
__global__ void box_stats_kernel(const float* __restrict__ boxes,
                                 const float* __restrict__ scales,
                                 const float* __restrict__ pdls,
                                 const float* __restrict__ pdts,
                                 float* __restrict__ cx, float* __restrict__ cy,
                                 float* __restrict__ tbw, float* __restrict__ tbh) {
  const int b = blockIdx.x;
  const int i = threadIdx.x;   // 0..511
  const float* bx = boxes + ((size_t)b * 512 + i) * 4;
  const float sc = scales[b], pl = pdls[b], pt = pdts[b];
  const float x1 = (bx[0] - pl) / sc, y1 = (bx[1] - pt) / sc;
  const float x2 = (bx[2] - pl) / sc, y2 = (bx[3] - pt) / sc;
  cx[b * 512 + i] = (x1 + x2) * 0.5f;
  cy[b * 512 + i] = (y1 + y2) * 0.5f;
  __shared__ float sw[512];
  __shared__ float sh[512];
  sw[i] = fmaxf(x1, x2);
  sh[i] = fmaxf(y1, y2);
  __syncthreads();
  for (int o = 256; o > 0; o >>= 1) {
    if (i < o) {
      sw[i] = fmaxf(sw[i], sw[i + o]);
      sh[i] = fmaxf(sh[i], sh[i + o]);
    }
    __syncthreads();
  }
  if (i == 0) { tbw[b] = sw[0]; tbh[b] = sh[0]; }
}

// ---------------------------------------------------------------------------
// 7) Pairwise attrs: histogram pass, threshold scan, selection pass.
//    Deterministic recompute; only the selected SET matters for segment_sum.
// ---------------------------------------------------------------------------
__device__ __forceinline__ void pair_decode(int p, int& i, int& j) {
  i = p / 511;
  int r = p - i * 511;
  j = r + (r >= i ? 1 : 0);
}

__global__ void attr_hist_kernel(const float* __restrict__ cx, const float* __restrict__ cy,
                                 const float* __restrict__ tbw, const float* __restrict__ tbh,
                                 int* __restrict__ hist) {
  int idx = blockIdx.x * blockDim.x + threadIdx.x;
  if (idx >= 4 * NPAIR) return;
  int b = idx / NPAIR;
  int p = idx - b * NPAIR;
  int i, j; pair_decode(p, i, j);
  float dy = (cy[b * 512 + i] - cy[b * 512 + j]) * 5.0f / tbh[b];
  float dx = (cx[b * 512 + i] - cx[b * 512 + j]) * 5.0f / tbw[b];
  float ar = expf(-dy * dy);
  float ac = expf(-dx * dx);
  int binr = min(1023, (int)(ar * 1024.0f));
  int binc = min(1023, (int)(ac * 1024.0f));
  atomicAdd(&hist[(0 * 4 + b) * 1024 + binr], 1);
  atomicAdd(&hist[(1 * 4 + b) * 1024 + binc], 1);
}

__global__ void find_thresh_kernel(const int* __restrict__ hist, int* __restrict__ thr) {
  int tb = threadIdx.x;
  if (tb >= 8) return;
  const int* h = hist + tb * 1024;
  int cum = 0, binSel = 0, above = 0;
  for (int bin = 1023; bin >= 0; --bin) {
    cum += h[bin];
    if (cum >= KEDGE) { binSel = bin; above = cum - h[bin]; break; }
  }
  thr[tb * 3 + 0] = binSel;
  thr[tb * 3 + 1] = above;
  thr[tb * 3 + 2] = KEDGE - above;
}

__global__ void select_edges_kernel(const float* __restrict__ cx, const float* __restrict__ cy,
                                    const float* __restrict__ tbw, const float* __restrict__ tbh,
                                    const int* __restrict__ thr, int* __restrict__ ctr,
                                    int* __restrict__ esrc, int* __restrict__ edst,
                                    float* __restrict__ ew) {
  int idx = blockIdx.x * blockDim.x + threadIdx.x;
  if (idx >= 4 * NPAIR) return;
  int b = idx / NPAIR;
  int p = idx - b * NPAIR;
  int i, j; pair_decode(p, i, j);
  float dy = (cy[b * 512 + i] - cy[b * 512 + j]) * 5.0f / tbh[b];
  float dx = (cx[b * 512 + i] - cx[b * 512 + j]) * 5.0f / tbw[b];
  float a2[2] = { expf(-dy * dy), expf(-dx * dx) };
#pragma unroll
  for (int type = 0; type < 2; ++type) {
    float a = a2[type];
    int bin = min(1023, (int)(a * 1024.0f));
    int tb = type * 4 + b;
    int tbin = thr[tb * 3 + 0];
    if (bin < tbin) continue;
    int slot;
    if (bin > tbin) {
      slot = atomicAdd(&ctr[tb * 2 + 0], 1);
    } else {
      int bs = atomicAdd(&ctr[tb * 2 + 1], 1);
      if (bs >= thr[tb * 3 + 2]) continue;
      slot = thr[tb * 3 + 1] + bs;
    }
    if (slot >= KEDGE) continue;
    int base = type * (4 * KEDGE) + b * KEDGE + slot;
    esrc[base] = b * 512 + i;
    edst[base] = b * 512 + j;
    ew[base] = a;
  }
}

// ---------------------------------------------------------------------------
// 8) GCN aggregation: deg (self=1 + edge weights), agg = D^-1/2 A D^-1/2 xw
// ---------------------------------------------------------------------------
__global__ void deg_init_kernel(float* __restrict__ deg) {
  int i = blockIdx.x * blockDim.x + threadIdx.x;
  if (i < 4096) deg[i] = 1.0f;
}

__global__ void deg_edge_kernel(const int* __restrict__ edst, const float* __restrict__ ew,
                                float* __restrict__ deg) {
  int e = blockIdx.x * blockDim.x + threadIdx.x;
  if (e >= 2 * 4 * KEDGE) return;
  int type = e / (4 * KEDGE);
  atomicAdd(&deg[type * 2048 + edst[e]], ew[e]);
}

__global__ void agg_self_kernel(const float* __restrict__ xw, const float* __restrict__ deg,
                                float* __restrict__ agg) {
  size_t i = (size_t)blockIdx.x * 256 + threadIdx.x;
  if (i >= (size_t)2 * 2048 * 512) return;
  agg[i] = xw[i] / deg[i >> 9];   // self-loop norm = 1/deg
}

__global__ void agg_edge_kernel(const int* __restrict__ esrc, const int* __restrict__ edst,
                                const float* __restrict__ ew, const float* __restrict__ deg,
                                const float* __restrict__ xw, float* __restrict__ agg) {
  const int type = blockIdx.y;
  const int e = blockIdx.x >> 1;
  const int c = ((blockIdx.x & 1) << 8) + threadIdx.x;
  const int ei = type * (4 * KEDGE) + e;
  const int s = esrc[ei], d = edst[ei];
  const float w = ew[ei];
  const float norm = rsqrtf(deg[type * 2048 + s]) * w * rsqrtf(deg[type * 2048 + d]);
  atomicAdd(&agg[((size_t)type * 2048 + d) * 512 + c],
            norm * xw[((size_t)type * 2048 + s) * 512 + c]);
}

__global__ void gcn_final_kernel(const float* __restrict__ agg,
                                 const float* __restrict__ bR, const float* __restrict__ bC,
                                 unsigned short* __restrict__ featB) {
  size_t i = (size_t)blockIdx.x * 256 + threadIdx.x;
  if (i >= (size_t)2 * 2048 * 512) return;
  int c = (int)(i & 511);
  int type = (int)(i >> 20);        // 2048*512 = 1<<20
  float v = agg[i] + (type ? bC[c] : bR[c]);
  featB[i] = f2bf(fmaxf(v, 0.0f));
}

// ---------------------------------------------------------------------------
// Host launcher
// ---------------------------------------------------------------------------
extern "C" void kernel_launch(void* const* d_in, const int* in_sizes, int n_in,
                              void* d_out, int out_size, void* d_ws, size_t ws_size,
                              hipStream_t stream) {
  (void)in_sizes; (void)n_in; (void)out_size; (void)ws_size;
  const float* input     = (const float*)d_in[0];
  const float* boxes     = (const float*)d_in[1];
  const float* im_scales = (const float*)d_in[2];
  const float* pdls      = (const float*)d_in[3];
  const float* pdts      = (const float*)d_in[4];
  const float* W_dec     = (const float*)d_in[5];
  const float* b_dec     = (const float*)d_in[6];
  const float* W_box     = (const float*)d_in[7];
  const float* b_box     = (const float*)d_in[8];
  const float* W_cnn     = (const float*)d_in[9];
  const float* b_cnn     = (const float*)d_in[10];
  const float* W_grow    = (const float*)d_in[11];
  const float* b_grow    = (const float*)d_in[12];
  const float* W_gcol    = (const float*)d_in[13];
  const float* b_gcol    = (const float*)d_in[14];
  const float* W_rcls    = (const float*)d_in[15];
  const float* b_rcls    = (const float*)d_in[16];
  const float* W_ccls    = (const float*)d_in[17];
  const float* b_ccls    = (const float*)d_in[18];
  float* out = (float*)d_out;

  char* ws = (char*)d_ws;
  size_t off = 0;
  auto alloc = [&](size_t bytes) -> void* {
    void* p = ws + off;
    off += (bytes + 255) & ~(size_t)255;
    return p;
  };

  unsigned short* inb    = (unsigned short*)alloc((size_t)9216 * 1024 * 2);  // NHWC bf16
  uint4*          wdecF  = (uint4*)alloc((size_t)288 * 16 * 64 * 16);
  float*          dec    = (float*)alloc((size_t)9216 * 256 * 4);            // NHWC f32
  unsigned short* cnnB   = (unsigned short*)alloc((size_t)2048 * 1024 * 2);
  uint4*          wcnnF  = (uint4*)alloc((size_t)32 * 32 * 64 * 16);
  unsigned short* fusion = (unsigned short*)alloc((size_t)2048 * 768 * 2);
  uint4*          wgrowF = (uint4*)alloc((size_t)24 * 32 * 64 * 16);
  uint4*          wgcolF = (uint4*)alloc((size_t)24 * 32 * 64 * 16);
  float*          xw     = (float*)alloc((size_t)2 * 2048 * 512 * 4);        // [type][node][c]
  float*          cxA    = (float*)alloc((size_t)2048 * 4);
  float*          cyA    = (float*)alloc((size_t)2048 * 4);
  float*          tbwA   = (float*)alloc((size_t)4 * 4);
  float*          tbhA   = (float*)alloc((size_t)4 * 4);
  int*            hist   = (int*)alloc((size_t)8 * 1024 * 4);
  int*            thr    = (int*)alloc((size_t)8 * 3 * 4);
  int*            ctr    = (int*)alloc((size_t)8 * 2 * 4);
  int*            esrc   = (int*)alloc((size_t)2 * 4 * KEDGE * 4);
  int*            edst   = (int*)alloc((size_t)2 * 4 * KEDGE * 4);
  float*          ewA    = (float*)alloc((size_t)2 * 4 * KEDGE * 4);
  float*          deg    = (float*)alloc((size_t)2 * 2048 * 4);
  float*          agg    = (float*)alloc((size_t)2 * 2048 * 512 * 4);
  unsigned short* featB  = (unsigned short*)alloc((size_t)2 * 2048 * 512 * 2);
  uint4*          wrclsF = (uint4*)alloc((size_t)16 * 15 * 64 * 16);
  uint4*          wcclsF = (uint4*)alloc((size_t)16 * 8 * 64 * 16);

  // Zero the buffers that need a clean slate every call (graph-capturable).
  hipMemsetAsync(hist, 0, (size_t)8 * 1024 * 4, stream);
  hipMemsetAsync(ctr, 0, (size_t)8 * 2 * 4, stream);
  hipMemsetAsync(esrc, 0, (size_t)2 * 4 * KEDGE * 4, stream);
  hipMemsetAsync(edst, 0, (size_t)2 * 4 * KEDGE * 4, stream);
  hipMemsetAsync(ewA, 0, (size_t)2 * 4 * KEDGE * 4, stream);

  // --- input + weight conditioning ---
  nchw_to_nhwc_bf16<<<36864, 256, 0, stream>>>(input, inb);
  preswz_kernel<<<(288 * 16 * 64) / 256, 256, 0, stream>>>(W_dec, wdecF, 288, 16, 9216, 256, 1);
  preswz_kernel<<<(32 * 32 * 64) / 256, 256, 0, stream>>>(W_cnn, wcnnF, 32, 32, 1024, 512, 0);
  preswz_kernel<<<(24 * 32 * 64) / 256, 256, 0, stream>>>(W_grow, wgrowF, 24, 32, 768, 512, 0);
  preswz_kernel<<<(24 * 32 * 64) / 256, 256, 0, stream>>>(W_gcol, wgcolF, 24, 32, 768, 512, 0);
  preswz_kernel<<<(16 * 15 * 64) / 256, 256, 0, stream>>>(W_rcls, wrclsF, 16, 15, 512, 228, 0);
  preswz_kernel<<<(16 * 8 * 64) / 256, 256, 0, stream>>>(W_ccls, wcclsF, 16, 8, 512, 116, 0);

  // --- decoder conv (WMMA implicit GEMM, 64M x 128N blocks) ---
  conv_wmma_kernel<<<dim3(144, 2), 256, 0, stream>>>(inb, wdecF, b_dec, dec);

  // --- ROI align + feature heads ---
  roi_kernel<<<2048, 256, 0, stream>>>(boxes, dec, cnnB);
  box_feat_kernel<<<2048, 256, 0, stream>>>(boxes, W_box, b_box, fusion);
  // cnn_feat = relu(cnn @ W_cnn + b_cnn) -> fusion[:, 256:768] (bf16)
  gemm_wmma_kernel<<<dim3(32, 4), 256, 0, stream>>>(cnnB, 1024, wcnnF, 32, b_cnn,
                                                    nullptr, fusion, 768, 256, 512, 32, 1);

  // --- edge attrs + top-k selection ---
  box_stats_kernel<<<4, 512, 0, stream>>>(boxes, im_scales, pdls, pdts, cxA, cyA, tbwA, tbhA);
  attr_hist_kernel<<<(4 * NPAIR) / 256, 256, 0, stream>>>(cxA, cyA, tbwA, tbhA, hist);
  find_thresh_kernel<<<1, 8, 0, stream>>>(hist, thr);
  select_edges_kernel<<<(4 * NPAIR) / 256, 256, 0, stream>>>(cxA, cyA, tbwA, tbhA,
                                                             thr, ctr, esrc, edst, ewA);

  // --- GCN ---
  deg_init_kernel<<<16, 256, 0, stream>>>(deg);
  deg_edge_kernel<<<(2 * 4 * KEDGE) / 256, 256, 0, stream>>>(edst, ewA, deg);
  // xw = fusion @ W_grow / W_gcol (f32, no bias yet)
  gemm_wmma_kernel<<<dim3(32, 4), 256, 0, stream>>>(fusion, 768, wgrowF, 32, nullptr,
                                                    xw, nullptr, 512, 0, 512, 24, 0);
  gemm_wmma_kernel<<<dim3(32, 4), 256, 0, stream>>>(fusion, 768, wgcolF, 32, nullptr,
                                                    xw + (size_t)2048 * 512, nullptr,
                                                    512, 0, 512, 24, 0);
  agg_self_kernel<<<8192, 256, 0, stream>>>(xw, deg, agg);
  agg_edge_kernel<<<dim3(2 * 4 * KEDGE, 2), 256, 0, stream>>>(esrc, edst, ewA,
                                                              deg, xw, agg);
  gcn_final_kernel<<<8192, 256, 0, stream>>>(agg, b_grow, b_gcol, featB);

  // --- classifiers (leaky ReLU 0.01) -> d_out ---
  gemm_wmma_kernel<<<dim3(32, 2), 256, 0, stream>>>(featB, 512, wrclsF, 15, b_rcls,
                                                    out, nullptr, 228, 0, 228, 16, 2);
  gemm_wmma_kernel<<<dim3(32, 1), 256, 0, stream>>>(featB + ((size_t)1 << 20), 512,
                                                    wcclsF, 8, b_ccls,
                                                    out + (size_t)2048 * 228, nullptr,
                                                    116, 0, 116, 16, 2);
}
